// TransformerMultiheadAttentionMechanism_64132451664197
// MI455X (gfx1250) — compile-verified, matrix operand order checked by reading, not run on hardware
//
#include <hip/hip_runtime.h>
#include <stdint.h>

// ---------------- CDNA5 WMMA types ----------------
typedef __attribute__((ext_vector_type(16))) __bf16 v16bf;
typedef __attribute__((ext_vector_type(4)))  __bf16 v4bf;
typedef __attribute__((ext_vector_type(8)))  float  v8f;

union FragBF {
  v16bf v;
  uint4 q[2];
};

__device__ __forceinline__ v8f wmma_bf16(v16bf a, v16bf b, v8f c) {
  return __builtin_amdgcn_wmma_f32_16x16x32_bf16(false, a, false, b, (short)0, c,
                                                 false, false);
}

// Fragment loader (A and B share the same per-lane pattern on CDNA5):
// lane L: row = row0 + (L&15); halves (L>>4) split K:
//   elems 0..7  <- cols col0 + (L>>4)*8 + 0..7
//   elems 8..15 <- cols col0 + 16 + (L>>4)*8 + 0..7
__device__ __forceinline__ v16bf load_frag_bf16(const __bf16* base, int rowStride,
                                                int row0, int col0, int lane) {
  const __bf16* p =
      base + (size_t)(row0 + (lane & 15)) * rowStride + col0 + (lane >> 4) * 8;
  FragBF f;
  f.q[0] = *(const uint4*)(p);
  f.q[1] = *(const uint4*)(p + 16);
  return f.v;
}

__device__ __forceinline__ v16bf load_frag_f32(const float* base, int rowStride,
                                               int row0, int col0, int lane) {
  const float* p =
      base + (size_t)(row0 + (lane & 15)) * rowStride + col0 + (lane >> 4) * 8;
  float4 a0 = *(const float4*)(p);
  float4 a1 = *(const float4*)(p + 4);
  float4 a2 = *(const float4*)(p + 16);
  float4 a3 = *(const float4*)(p + 20);
  v16bf f;
  f[0]  = (__bf16)a0.x; f[1]  = (__bf16)a0.y; f[2]  = (__bf16)a0.z; f[3]  = (__bf16)a0.w;
  f[4]  = (__bf16)a1.x; f[5]  = (__bf16)a1.y; f[6]  = (__bf16)a1.z; f[7]  = (__bf16)a1.w;
  f[8]  = (__bf16)a2.x; f[9]  = (__bf16)a2.y; f[10] = (__bf16)a2.z; f[11] = (__bf16)a2.w;
  f[12] = (__bf16)a3.x; f[13] = (__bf16)a3.y; f[14] = (__bf16)a3.z; f[15] = (__bf16)a3.w;
  return f;
}

// Problem constants
constexpr int Bc = 4, Tc = 2048, Dc = 512, Hc = 8, DKc = 64;

// ---------------- K1: fused QKV projection (fp32 -> bf16, V transposed) ----
// z=0: K = key@Wk^T+bk   -> Kb row-major [B*T, D] bf16
// z=1: Q = query@Wq^T+bq -> Qb row-major [B*T, D] bf16
// z=2: V = value@Wv^T+bv -> Vt transposed [(b*D + n)][t] bf16
__global__ __launch_bounds__(128) void qkv_proj_kernel(
    const float* __restrict__ key, const float* __restrict__ value,
    const float* __restrict__ query,
    const float* __restrict__ Wk, const float* __restrict__ bk,
    const float* __restrict__ Wv, const float* __restrict__ bv,
    const float* __restrict__ Wq, const float* __restrict__ bq,
    __bf16* __restrict__ Kb, __bf16* __restrict__ Qb, __bf16* __restrict__ Vt) {
  const int lane = threadIdx.x & 31;
  const int wv = threadIdx.x >> 5;
  const int n0 = (blockIdx.x * 4 + wv) * 16;
  const int m0 = blockIdx.y * 16;
  const int which = blockIdx.z;

  const float *src, *W, *bias;
  if (which == 0)      { src = key;   W = Wk; bias = bk; }
  else if (which == 1) { src = query; W = Wq; bias = bq; }
  else                 { src = value; W = Wv; bias = bv; }

  v8f c0 = {}, c1 = {};
#pragma unroll 2
  for (int k0 = 0; k0 < Dc; k0 += 64) {
    c0 = wmma_bf16(load_frag_f32(src, Dc, m0, k0, lane),
                   load_frag_f32(W,   Dc, n0, k0, lane), c0);
    c1 = wmma_bf16(load_frag_f32(src, Dc, m0, k0 + 32, lane),
                   load_frag_f32(W,   Dc, n0, k0 + 32, lane), c1);
  }
  v8f c = c0 + c1;
  const int half = lane >> 4;
  const int n = n0 + (lane & 15);
  const float bn = bias[n];
  if (which == 2) {
    const int bb = m0 >> 11;         // batch
    const int t = m0 & (Tc - 1);     // row within batch
#pragma unroll
    for (int r = 0; r < 8; r++) {
      Vt[(size_t)(bb * Dc + n) * Tc + (t + r + 8 * half)] = (__bf16)(c[r] + bn);
    }
  } else {
    __bf16* out = (which == 0) ? Kb : Qb;
#pragma unroll
    for (int r = 0; r < 8; r++) {
      out[(size_t)(m0 + r + 8 * half) * Dc + n] = (__bf16)(c[r] + bn);
    }
  }
}

// ---------------- K2: fused attention per (b, h, 16-row q tile) -----------
// 4 waves / block. LDS: E fp32 [16][2052], Abf bf16 [16][2056], stats.
__global__ __launch_bounds__(128) void attn_kernel(
    const __bf16* __restrict__ Qb, const __bf16* __restrict__ Kb,
    const __bf16* __restrict__ Vt, const int* __restrict__ mask,
    float* __restrict__ aw, __bf16* __restrict__ Cb) {
  constexpr int ES = 2052;  // energy row stride (floats), bank-conflict padded
  constexpr int AS = 2056;  // prob row stride (bf16)
  extern __shared__ __attribute__((aligned(16))) unsigned char smem[];
  float* E = (float*)smem;
  __bf16* Abf = (__bf16*)(smem + 16 * ES * 4);
  float* wstat = (float*)(smem + 16 * ES * 4 + 16 * AS * 2);
  float* wm = wstat;        // [4][16]
  float* wl = wstat + 64;   // [4][16]
  float* Mr = wstat + 128;  // [16]
  float* Lr = wstat + 144;  // [16]

  const int lane = threadIdx.x & 31;
  const int wv = threadIdx.x >> 5;
  const int laneN = lane & 15;
  const int half = lane >> 4;
  const int tq0 = blockIdx.x * 16;
  const int h = blockIdx.y;
  const int bb = blockIdx.z;
  const int colH = h * DKc;

  // ---- Phase 1: energy tiles -> LDS, online row max/sum per wave ----
  const int qr0 = bb * Tc + tq0;
  v16bf aQ0 = load_frag_bf16(Qb, Dc, qr0, colH, lane);
  v16bf aQ1 = load_frag_bf16(Qb, Dc, qr0, colH + 32, lane);

  float mrun[8], lrun[8];
#pragma unroll
  for (int r = 0; r < 8; r++) { mrun[r] = -1e30f; lrun[r] = 0.0f; }

  const float scale = 0.125f;  // 1/sqrt(64)
  const int* mrow = mask + ((size_t)bb * Tc + tq0) * Tc;

  // each wave: 32-col block pairs, 4 WMMAs in 2 independent accumulator chains
  for (int p = wv; p < Tc / 32; p += 4) {
    const int tk0 = p * 32;
    const int kr = bb * Tc + tk0;
    // prefetch next iteration: one 128B line per lane covers all 4 K frags,
    // one line per lane covers the next 16x32 mask tile (global_prefetch_b8)
    {
      const int pn = (p + 4 < Tc / 32) ? p + 4 : p;
      const int tkn = pn * 32;
      __builtin_prefetch(Kb + (size_t)(bb * Tc + tkn + laneN + half * 16) * Dc + colH, 0, 1);
      __builtin_prefetch(&mrow[(size_t)laneN * Tc + tkn + half * 16], 0, 1);
    }
    v16bf bK0a = load_frag_bf16(Kb, Dc, kr,      colH,      lane);
    v16bf bK0b = load_frag_bf16(Kb, Dc, kr,      colH + 32, lane);
    v16bf bK1a = load_frag_bf16(Kb, Dc, kr + 16, colH,      lane);
    v16bf bK1b = load_frag_bf16(Kb, Dc, kr + 16, colH + 32, lane);
    v8f c0 = {}, c1 = {};
    c0 = wmma_bf16(aQ0, bK0a, c0);
    c1 = wmma_bf16(aQ0, bK1a, c1);
    c0 = wmma_bf16(aQ1, bK0b, c0);
    c1 = wmma_bf16(aQ1, bK1b, c1);
#pragma unroll
    for (int r = 0; r < 8; r++) {
      const int m = r + 8 * half;   // C layout: VGPR r -> rows r / r+8
      float e0 = c0[r] * scale;
      float e1 = c1[r] * scale;
      if (mrow[(size_t)m * Tc + tk0 + laneN] == 0)      e0 = -1024.0f;
      if (mrow[(size_t)m * Tc + tk0 + 16 + laneN] == 0) e1 = -1024.0f;
      E[m * ES + tk0 + laneN] = e0;
      E[m * ES + tk0 + 16 + laneN] = e1;
      // reduce over the 16 lanes of this half (xor 1,2,4,8 stay in-half)
      float bm = fmaxf(e0, e1);
      bm = fmaxf(bm, __shfl_xor(bm, 1, 32));
      bm = fmaxf(bm, __shfl_xor(bm, 2, 32));
      bm = fmaxf(bm, __shfl_xor(bm, 4, 32));
      bm = fmaxf(bm, __shfl_xor(bm, 8, 32));
      const float nm = fmaxf(mrun[r], bm);
      float t = __expf(e0 - nm) + __expf(e1 - nm);
      t += __shfl_xor(t, 1, 32);
      t += __shfl_xor(t, 2, 32);
      t += __shfl_xor(t, 4, 32);
      t += __shfl_xor(t, 8, 32);
      lrun[r] = lrun[r] * __expf(mrun[r] - nm) + t;
      mrun[r] = nm;
    }
  }
  if (laneN == 0) {
#pragma unroll
    for (int r = 0; r < 8; r++) {
      wm[wv * 16 + r + 8 * half] = mrun[r];
      wl[wv * 16 + r + 8 * half] = lrun[r];
    }
  }
  __syncthreads();
  if (threadIdx.x < 16) {
    const int row = threadIdx.x;
    float M = -1e30f;
    for (int w2 = 0; w2 < 4; w2++) M = fmaxf(M, wm[w2 * 16 + row]);
    float L = 0.0f;
    for (int w2 = 0; w2 < 4; w2++) L += wl[w2 * 16 + row] * __expf(wm[w2 * 16 + row] - M);
    Mr[row] = M;
    Lr[row] = L;
  }
  __syncthreads();

  // ---- Phase 2: normalize; aw fp32 -> HBM once (coalesced), bf16 -> LDS ----
  float* awt = aw + (((size_t)bb * Hc + h) * Tc + tq0) * Tc;
  for (int rr = 0; rr < 4; rr++) {
    const int row = wv * 4 + rr;
    const float M = Mr[row];
    const float invL = 1.0f / Lr[row];
#pragma unroll 4
    for (int it = 0; it < 16; it++) {
      const int col = it * 128 + lane * 4;
      float4 e4 = *(float4*)&E[row * ES + col];
      float4 p;
      p.x = __expf(e4.x - M) * invL;
      p.y = __expf(e4.y - M) * invL;
      p.z = __expf(e4.z - M) * invL;
      p.w = __expf(e4.w - M) * invL;
      *(float4*)&awt[(size_t)row * Tc + col] = p;
      v4bf pb;
      pb[0] = (__bf16)p.x; pb[1] = (__bf16)p.y;
      pb[2] = (__bf16)p.z; pb[3] = (__bf16)p.w;
      *(v4bf*)&Abf[row * AS + col] = pb;
    }
  }
  __syncthreads();

  // ---- Phase 3: context tile = aw(bf16,LDS) @ V(bf16,transposed) ----
  const int n0 = wv * 16;  // 4 waves cover DK=64
  const int vr = bb * Dc + colH + n0;
  v8f c0 = {}, c1 = {};
  for (int kb = 0; kb < Tc / 64; kb++) {
    const int tk0 = kb * 64;
    // prefetch next iteration's V rows (64 contiguous bf16 = one 128B line)
    {
      const int tkn = (tk0 + 64 < Tc) ? tk0 + 64 : tk0;
      __builtin_prefetch(Vt + (size_t)(vr + laneN) * Tc + tkn, 0, 1);
    }
    FragBF a0, a1;
    const __bf16* ap = &Abf[laneN * AS + tk0 + half * 8];
    a0.q[0] = *(const uint4*)ap;
    a0.q[1] = *(const uint4*)(ap + 16);
    a1.q[0] = *(const uint4*)(ap + 32);
    a1.q[1] = *(const uint4*)(ap + 48);
    v16bf bV0 = load_frag_bf16(Vt, Tc, vr, tk0, lane);
    v16bf bV1 = load_frag_bf16(Vt, Tc, vr, tk0 + 32, lane);
    c0 = wmma_bf16(a0.v, bV0, c0);
    c1 = wmma_bf16(a1.v, bV1, c1);
  }
  v8f c = c0 + c1;
#pragma unroll
  for (int r = 0; r < 8; r++) {
    const int m = tq0 + r + 8 * half;
    Cb[((size_t)bb * Tc + m) * Dc + colH + n0 + laneN] = (__bf16)c[r];
  }
}

// ---------------- K3: output projection (bf16 A, fp32 W -> fp32 + bias) ---
__global__ __launch_bounds__(128) void out_proj_kernel(
    const __bf16* __restrict__ Cb, const float* __restrict__ Wo,
    const float* __restrict__ bo, float* __restrict__ out) {
  const int lane = threadIdx.x & 31;
  const int wv = threadIdx.x >> 5;
  const int n0 = (blockIdx.x * 4 + wv) * 16;
  const int m0 = blockIdx.y * 16;
  v8f c0 = {}, c1 = {};
#pragma unroll 2
  for (int k0 = 0; k0 < Dc; k0 += 64) {
    c0 = wmma_bf16(load_frag_bf16(Cb, Dc, m0, k0, lane),
                   load_frag_f32(Wo, Dc, n0, k0, lane), c0);
    c1 = wmma_bf16(load_frag_bf16(Cb, Dc, m0, k0 + 32, lane),
                   load_frag_f32(Wo, Dc, n0, k0 + 32, lane), c1);
  }
  v8f c = c0 + c1;
  const int half = lane >> 4;
  const int n = n0 + (lane & 15);
  const float bn = bo[n];
#pragma unroll
  for (int r = 0; r < 8; r++) {
    out[(size_t)(m0 + r + 8 * half) * Dc + n] = c[r] + bn;
  }
}

// ---------------- host launcher -------------------------------------------
extern "C" void kernel_launch(void* const* d_in, const int* in_sizes, int n_in,
                              void* d_out, int out_size, void* d_ws, size_t ws_size,
                              hipStream_t stream) {
  (void)in_sizes; (void)n_in; (void)out_size; (void)ws_size;
  const float* key   = (const float*)d_in[0];
  const float* value = (const float*)d_in[1];
  const float* query = (const float*)d_in[2];
  const int*   mask  = (const int*)d_in[3];
  const float* Wk = (const float*)d_in[4];
  const float* bk = (const float*)d_in[5];
  const float* Wv = (const float*)d_in[6];
  const float* bv = (const float*)d_in[7];
  const float* Wq = (const float*)d_in[8];
  const float* bq = (const float*)d_in[9];
  const float* Wo = (const float*)d_in[10];
  const float* bo = (const float*)d_in[11];

  float* ctx_out = (float*)d_out;
  float* aw_out = ctx_out + (size_t)Bc * Tc * Dc;

  const size_t mat = (size_t)Bc * Tc * Dc;  // 4.19M elems
  __bf16* Kb = (__bf16*)d_ws;
  __bf16* Qb = Kb + mat;
  __bf16* Vt = Qb + mat;
  __bf16* Cb = Vt + mat;

  dim3 blk(128);
  qkv_proj_kernel<<<dim3(Dc / 64, (Bc * Tc) / 16, 3), blk, 0, stream>>>(
      key, value, query, Wk, bk, Wv, bv, Wq, bq, Kb, Qb, Vt);

  const size_t smem = 16 * 2052 * 4 + 16 * 2056 * 2 + 160 * 4;  // ~198 KB
  attn_kernel<<<dim3(Tc / 16, Hc, Bc), blk, smem, stream>>>(Qb, Kb, Vt, mask,
                                                            aw_out, Cb);

  out_proj_kernel<<<dim3(Dc / 64, (Bc * Tc) / 16), blk, 0, stream>>>(Cb, Wo, bo,
                                                                     ctx_out);
}
